// Decoder_8718783611480
// MI455X (gfx1250) — matrix operand order repaired and checked by reading, not compiled, not run on hardware
//
#include <hip/hip_runtime.h>
#include <hip/hip_bf16.h>

// Problem constants (from reference)
#define B_   2048
#define T_   240
#define D_   90
#define NCLS 40
#define TC   16      // t-chunks
#define TPC  15      // timesteps per chunk (240/16)

typedef __attribute__((ext_vector_type(16))) __bf16 v16bf;
typedef __attribute__((ext_vector_type(8)))  __bf16 v8bf;
typedef __attribute__((ext_vector_type(8)))  float  v8f;
typedef __attribute__((ext_vector_type(2)))  float  v2f;

// Workspace layout (bytes)
#define WIH_OFF   0                          // 3 gates * 3 Ksteps * 6 Nslices * 512 bf16 = 55296 B
#define WOUT_OFF  55296                      // 240 t * 3 Ksteps * 3 Ntiles * 512 bf16 = 2211840 B
#define PART_OFF  (WOUT_OFF + 2211840)       // TC * B * 40 f32 = 5242880 B

#define LOG2E  1.442695041f

// tanh: prefer CDNA5 hardware v_tanh_f32
__device__ __forceinline__ float fast_tanh(float x) {
#if __has_builtin(__builtin_amdgcn_tanhf)
  return __builtin_amdgcn_tanhf(x);
#elif __has_builtin(__builtin_amdgcn_tanh_f32)
  return __builtin_amdgcn_tanh_f32(x);
#else
  float e = __builtin_amdgcn_exp2f(x * (2.0f * LOG2E));
  return 1.0f - 2.0f * __builtin_amdgcn_rcpf(e + 1.0f);
#endif
}

// sigmoid(x) = 0.5 + 0.5*tanh(x/2): 3 VALU ops with hardware tanh
__device__ __forceinline__ float sigf(float x) {
#if __has_builtin(__builtin_amdgcn_tanhf) || __has_builtin(__builtin_amdgcn_tanh_f32)
  return fmaf(0.5f, fast_tanh(x * 0.5f), 0.5f);
#else
  return __builtin_amdgcn_rcpf(1.0f + __builtin_amdgcn_exp2f(x * -LOG2E));
#endif
}

// ---------------------------------------------------------------------------
// Prep: swizzle weights into WMMA fragment-major bf16 layout (contiguous k order).
// K padded 90->96; row k==90 of the gates' B matrix carries the combined bias
// (b_ih+b_hh) so the bias-add rides the GEMM (x fragment has 1.0 at k=90).
// B-fragment element q of lane l within a 32x16 tile is B[k,n] with
//   k = 32*s + (l<16 ? 0 : 16) + q,  n = 16*tile + (l & 15)
// ---------------------------------------------------------------------------
__global__ void prep_kernel(const float* __restrict__ W_ih,
                            const float* __restrict__ b_ih,
                            const float* __restrict__ b_hh,
                            const float* __restrict__ W_out,
                            __bf16* __restrict__ wih_frag,
                            __bf16* __restrict__ wout_frag)
{
  const int stride = gridDim.x * blockDim.x;
  const int tid = blockIdx.x * blockDim.x + threadIdx.x;

  // W_ih fragments: gates i(rows 0..89), g(180..269), o(270..359); f-gate dead.
  for (int i = tid; i < 3*3*6*512; i += stride) {
    int q    = i & 15;
    int lane = (i >> 4) & 31;
    int j    = (i >> 9) % 6;          // N-slice
    int s    = (i / 3072) % 3;        // K-step
    int ga   = i / 9216;              // gate: 0=i, 1=g, 2=o
    int gb   = (ga == 0) ? 0 : (ga == 1 ? 180 : 270);
    int k = 32*s + ((lane < 16) ? 0 : 16) + q;
    int n = 16*j + (lane & 15);
    float v = 0.0f;
    if (n < D_) {
      if (k < D_)       v = W_ih[(size_t)(gb + n) * D_ + k];
      else if (k == D_) v = b_ih[gb + n] + b_hh[gb + n];   // bias row
    }
    wih_frag[i] = (__bf16)v;
  }

  // W_out fragments per timestep: B2[k,n] = W_out[n, t*90 + k]
  for (int i = tid; i < 240*9*512; i += stride) {
    int q    = i & 15;
    int lane = (i >> 4) & 31;
    int n2   = (i >> 9) % 3;
    int s    = (i / 1536) % 3;
    int t    = i / 4608;
    int k = 32*s + ((lane < 16) ? 0 : 16) + q;
    int n = 16*n2 + (lane & 15);
    float v = 0.0f;
    if (k < D_ && n < NCLS) v = W_out[(size_t)n * (T_*D_) + t*D_ + k];
    wout_frag[i] = (__bf16)v;
  }
}

// ---------------------------------------------------------------------------
// Fused gates-GEMM + activations + output-GEMM.
// One wave per block; wave owns 16 batch rows x one t-chunk (15 steps).
// N-slice loop is deliberately NOT unrolled: keeps live B-tiles to 9 and
// VGPR pressure low enough for multi-wave occupancy (round-2 full unroll
// hoisted all 54 weight tiles -> ~680 VGPRs, 1 wave/SIMD).
// ---------------------------------------------------------------------------
__global__ __launch_bounds__(32) void lstm_wmma_kernel(
    const float*  __restrict__ x,
    const __bf16* __restrict__ wih_frag,
    const __bf16* __restrict__ wout_frag,
    float*        __restrict__ partials)
{
  const int lane  = threadIdx.x;              // 0..31
  const int m     = lane & 15;                // row within 16-tile (A/B column id)
  const int hi    = (threadIdx.x >> 4) & 1;   // lane half, provably 0/1
  const int b0    = blockIdx.x * 16;          // batch tile base
  const int chunk = blockIdx.y;               // t-chunk

  __shared__ __bf16 hbuf[16 * 96];            // h tile, rows x padded D

  v8f c_out[3];
  #pragma unroll
  for (int n2 = 0; n2 < 3; ++n2)
    #pragma unroll
    for (int r = 0; r < 8; ++r) c_out[n2][r] = 0.0f;

  const float* xrow = x + (size_t)(b0 + m) * T_ * D_;

  #pragma unroll 1
  for (int tt = 0; tt < TPC; ++tt) {
    const int t = chunk * TPC + tt;
    const float* xp = xrow + (size_t)t * D_;

    // ---- A fragments from x (f32 -> bf16), contiguous-k 16x32 A layout (matches B prep):
    // lane element q: K = 32*s + (hi?16:0) + q  -> two runs of 8.
    // Branchless OOB: clamp address to 88 and select {x, 1.0 (bias lane k=90), 0}.
    v16bf a[3];
    #pragma unroll
    for (int s = 0; s < 3; ++s) {
      const int d0 = 32*s + hi*16;
      #pragma unroll
      for (int p = 0; p < 4; ++p) {
        int d  = d0 + 2*p;
        int dc = (d < 88) ? d : 88;
        v2f u  = *(const v2f*)(xp + dc);
        a[s][2*p]   = (__bf16)(d   < D_ ? u.x : (d   == D_ ? 1.0f : 0.0f));
        a[s][2*p+1] = (__bf16)(d+1 < D_ ? u.y : (d+1 == D_ ? 1.0f : 0.0f));

        int d2  = d0 + 8 + 2*p;
        int dc2 = (d2 < 88) ? d2 : 88;
        v2f u2  = *(const v2f*)(xp + dc2);
        a[s][8+2*p]   = (__bf16)(d2   < D_ ? u2.x : (d2   == D_ ? 1.0f : 0.0f));
        a[s][8+2*p+1] = (__bf16)(d2+1 < D_ ? u2.y : (d2+1 == D_ ? 1.0f : 0.0f));
      }
    }

    // ---- Gates (i, g, o) per 16-wide N-slice, fused activations -> LDS.
    #pragma unroll 1
    for (int j = 0; j < 6; ++j) {
      v8f acc[3];
      #pragma unroll
      for (int ga = 0; ga < 3; ++ga) {
        v8f c;
        #pragma unroll
        for (int r = 0; r < 8; ++r) c[r] = 0.0f;
        #pragma unroll
        for (int s = 0; s < 3; ++s) {
          const v16bf bfrag =
              *(const v16bf*)(wih_frag + (((((ga*3 + s)*6) + j)*32 + lane) << 4));
          c = __builtin_amdgcn_wmma_f32_16x16x32_bf16(
                  false, a[s], false, bfrag, (short)0, c, false, false);
        }
        acc[ga] = c;
      }
      const int d = 16*j + m;   // C layout: N = lane&15, M = r + hi*8
      #pragma unroll
      for (int r = 0; r < 8; ++r) {
        float cc = sigf(acc[0][r]) * fast_tanh(acc[1][r]);   // f*c0 == 0
        float hh = sigf(acc[2][r]) * fast_tanh(cc);
        hbuf[(r + hi*8)*96 + d] = (__bf16)hh;                // pad cols -> h=0
      }
    }

    // Cross-lane LDS visibility within the wave (single-wave block, no barrier)
    asm volatile("s_wait_dscnt 0x0" ::: "memory");

    // ---- Re-pack h into A fragments via LDS (two ds_load_b128 per K-step)
    v16bf a2[3];
    #pragma unroll
    for (int s = 0; s < 3; ++s) {
      const int d0 = 32*s + hi*16;
      v8bf lo  = *(const v8bf*)(&hbuf[m*96 + d0]);
      v8bf hi8 = *(const v8bf*)(&hbuf[m*96 + d0 + 8]);
      a2[s] = __builtin_shufflevector(lo, hi8, 0,1,2,3,4,5,6,7,8,9,10,11,12,13,14,15);
    }

    // ---- Output GEMM: accumulate logits across the t-chunk
    #pragma unroll
    for (int n2 = 0; n2 < 3; ++n2) {
      #pragma unroll
      for (int s = 0; s < 3; ++s) {
        const v16bf bfrag =
            *(const v16bf*)(wout_frag + ((((((size_t)t*3 + s)*3) + n2)*32 + lane) << 4));
        c_out[n2] = __builtin_amdgcn_wmma_f32_16x16x32_bf16(
                        false, a2[s], false, bfrag, (short)0, c_out[n2], false, false);
      }
    }
    asm volatile("" ::: "memory");  // keep next-iter LDS stores after these reads
  }

  // ---- Emit per-chunk partial logits (bitwise-deterministic; no atomics)
  #pragma unroll
  for (int n2 = 0; n2 < 3; ++n2) {
    const int n = 16*n2 + m;
    if (n < NCLS) {
      #pragma unroll
      for (int r = 0; r < 8; ++r) {
        partials[((size_t)chunk * B_ + (b0 + r + hi*8)) * NCLS + n] = c_out[n2][r];
      }
    }
  }
}

// ---------------------------------------------------------------------------
// Reduce chunk partials, add b_out, softmax over groups of 10.
// ---------------------------------------------------------------------------
__global__ void softmax_kernel(const float* __restrict__ partials,
                               const float* __restrict__ b_out,
                               float* __restrict__ out)
{
  int idx = blockIdx.x * blockDim.x + threadIdx.x;
  if (idx >= B_ * 4) return;
  int b = idx >> 2, grp = idx & 3;
  float v[10];
  float mx = -3.4e38f;
  #pragma unroll
  for (int k2 = 0; k2 < 10; ++k2) {
    int n = grp*10 + k2;
    float s = b_out[n];
    #pragma unroll 1
    for (int c = 0; c < TC; ++c) s += partials[(size_t)c * B_ * NCLS + (size_t)b * NCLS + n];
    v[k2] = s;
    mx = fmaxf(mx, s);
  }
  float sum = 0.0f;
  #pragma unroll
  for (int k2 = 0; k2 < 10; ++k2) {
    v[k2] = __builtin_amdgcn_exp2f((v[k2] - mx) * LOG2E);
    sum += v[k2];
  }
  float inv = 1.0f / sum;
  #pragma unroll
  for (int k2 = 0; k2 < 10; ++k2) out[(size_t)b * NCLS + grp*10 + k2] = v[k2] * inv;
}

// ---------------------------------------------------------------------------
extern "C" void kernel_launch(void* const* d_in, const int* in_sizes, int n_in,
                              void* d_out, int out_size, void* d_ws, size_t ws_size,
                              hipStream_t stream)
{
  const float* x     = (const float*)d_in[0];
  const float* W_ih  = (const float*)d_in[1];
  // d_in[2] = W_hh: unused (h0 == 0 -> W_hh @ h0 == 0)
  const float* b_ih  = (const float*)d_in[3];
  const float* b_hh  = (const float*)d_in[4];
  const float* W_out = (const float*)d_in[5];
  const float* b_out = (const float*)d_in[6];

  char*   ws        = (char*)d_ws;
  __bf16* wih_frag  = (__bf16*)(ws + WIH_OFF);
  __bf16* wout_frag = (__bf16*)(ws + WOUT_OFF);
  float*  partials  = (float*)(ws + PART_OFF);

  prep_kernel<<<1024, 256, 0, stream>>>(W_ih, b_ih, b_hh, W_out,
                                        wih_frag, wout_frag);
  lstm_wmma_kernel<<<dim3(B_/16, TC), 32, 0, stream>>>(x, wih_frag, wout_frag,
                                                       partials);
  softmax_kernel<<<(B_*4 + 127)/128, 128, 0, stream>>>(partials, b_out, (float*)d_out);
}
